// gat_all_55422257988365
// MI455X (gfx1250) — compile-verified, matrix operand order checked by reading, not run on hardware
//
#include <hip/hip_runtime.h>
#include <hip/hip_bf16.h>

// ---------------------------------------------------------------------------
// GAT forward for MI455X (gfx1250):
//   - WMMA f16 GEMMs with double-buffered GLOBAL_LOAD_ASYNC_TO_LDS_B128 staging
//   - bandwidth-bound edge softmax/aggregation with native float atomics
// ---------------------------------------------------------------------------

typedef __attribute__((ext_vector_type(16))) _Float16 v16h;
typedef __attribute__((ext_vector_type(8)))  float    v8f;

#define N_NODES 50000
#define N_EDGES 800000
#define NFEAT   512
#define NHID    64
#define NHEAD   8
#define NCLASS  40
#define HCAT    (NHEAD * NHID)   // 512 (also the K-dim of both GEMMs)

// ---------------- float-atomic-max key encoding (order preserving) ----------
__device__ __forceinline__ unsigned int float_to_key(float f) {
  unsigned int u = __float_as_uint(f);
  return (u & 0x80000000u) ? ~u : (u | 0x80000000u);
}
__device__ __forceinline__ float key_to_float(unsigned int k) {
  unsigned int u = (k & 0x80000000u) ? (k & 0x7fffffffu) : ~k;
  return __uint_as_float(u);
}
__device__ __forceinline__ float lrelu(float v) { return v > 0.f ? v : 0.2f * v; }

// ---------------- prep kernels ---------------------------------------------
__global__ void gat_f32_to_f16(const float* __restrict__ in, _Float16* __restrict__ out, long count) {
  long i = (long)blockIdx.x * blockDim.x + threadIdx.x;
  if (i < count) out[i] = (_Float16)in[i];
}

// Wt1[h][n][k] = (f16) W1[h][k][n]   (h:8, k:512, n:64)
__global__ void gat_prep_w1(const float* __restrict__ W1, _Float16* __restrict__ Wt1) {
  int i = blockIdx.x * blockDim.x + threadIdx.x;          // < 8*64*512
  if (i >= NHEAD * NHID * NFEAT) return;
  int k = i & (NFEAT - 1);
  int n = (i >> 9) & (NHID - 1);
  int h = i >> 15;
  Wt1[i] = (_Float16)W1[(size_t)h * NFEAT * NHID + (size_t)k * NHID + n];
}

// Wt2[n][k] = (f16) W2[k][n], n padded 40 -> 64 with zeros   (k:512)
__global__ void gat_prep_w2(const float* __restrict__ W2, _Float16* __restrict__ Wt2) {
  int i = blockIdx.x * blockDim.x + threadIdx.x;          // < 64*512
  if (i >= 64 * NFEAT) return;
  int k = i & (NFEAT - 1);
  int n = i >> 9;
  Wt2[i] = (n < NCLASS) ? (_Float16)W2[(size_t)k * NCLASS + n] : (_Float16)0.f;
}

// ---------------- init kernels ---------------------------------------------
__global__ void gat_zero_md(unsigned int* mk, float* dn, int count) {
  int i = blockIdx.x * blockDim.x + threadIdx.x;
  if (i < count) { mk[i] = 0u; dn[i] = 0.f; }
}

__global__ void gat_init_bias(float* o, const float* __restrict__ b, long count, int C) {
  long i = (long)blockIdx.x * blockDim.x + threadIdx.x;
  if (i < count) o[i] = b[i % C];
}

// ---------------------------------------------------------------------------
// Unified WMMA GEMM:  Out[row, colbase + c] = sum_k Ah[row,k] * Bt[head][c][k]
//   Ah : [N_NODES][512] f16 row-major
//   Bt : [gridDim.y][64][512] f16 (column-of-B per row, i.e. pre-transposed)
//   A and B K-slabs are streamed into LDS with GLOBAL_LOAD_ASYNC_TO_LDS_B128,
//   double buffered, retired with s_wait_asynccnt (ASYNCcnt).
// ---------------------------------------------------------------------------
template <int NTILES>
__global__ __launch_bounds__(256) void gat_gemm_async_wmma(
    const _Float16* __restrict__ Ah, const _Float16* __restrict__ Bt,
    float* __restrict__ Out, int ldout, int colstride, int ncols) {
  __shared__ _Float16 As[2][128][32];   // [buf][m][k]
  __shared__ _Float16 Bs[2][64][32];    // [buf][n][k]

  const long rowBase = (long)blockIdx.x * 128;
  const int t    = threadIdx.x;
  const int wave = t >> 5;
  const int lane = t & 31;
  const int nlow = lane & 15;
  const int kb   = (lane >= 16) ? 8 : 0;    // ISA 16-bit A/B lane layout
  const int mrow = wave * 16 + nlow;
  const int colbase = blockIdx.y * colstride;

  const _Float16* __restrict__ Bp = Bt + (size_t)blockIdx.y * 64 * NFEAT;

  v8f acc[NTILES] = {};

  // --- async slab issue: per wave exactly 3 ASYNC issues (A:2, B:1) ---------
  auto issueA = [&](int k0, int b) {
    int r  = t >> 1;
    int kk = (t & 1) * 16;
    long grow = rowBase + r;
    if (grow >= N_NODES) grow = N_NODES - 1;      // clamp: pad rows read valid data
    unsigned long long ga = (unsigned long long)(size_t)(Ah + grow * (long)NFEAT + k0 + kk);
    unsigned lo = (unsigned)(size_t)&As[b][r][kk];
    asm volatile("global_load_async_to_lds_b128 %0, %1, off"
                 :: "v"(lo), "v"(ga) : "memory");
    asm volatile("global_load_async_to_lds_b128 %0, %1, off offset:16"
                 :: "v"(lo), "v"(ga) : "memory");
  };
  auto issueB = [&](int k0, int b) {
    int n = t >> 2;
    int q = t & 3;
    unsigned long long ga = (unsigned long long)(size_t)(Bp + (size_t)n * NFEAT + k0 + q * 8);
    unsigned lo = (unsigned)(size_t)&Bs[b][n][q * 8];
    asm volatile("global_load_async_to_lds_b128 %0, %1, off"
                 :: "v"(lo), "v"(ga) : "memory");
  };

  issueA(0, 0);
  issueB(0, 0);

  const int KSTEPS = NFEAT / 32;   // 16
  for (int ks = 0; ks < KSTEPS; ks++) {
    const int buf = ks & 1;
    const bool hasNext = (ks + 1) < KSTEPS;
    if (hasNext) {                         // prefetch next slab into other buffer
      issueA((ks + 1) * 32, buf ^ 1);
      issueB((ks + 1) * 32, buf ^ 1);
      asm volatile("s_wait_asynccnt 0x3" ::: "memory");   // oldest 3 (this slab) done
    } else {
      asm volatile("s_wait_asynccnt 0x0" ::: "memory");
    }
    __syncthreads();                       // all waves' slab data visible

    v16h afrag;
    {
      const _Float16* p0 = &As[buf][mrow][kb];
      const _Float16* p1 = &As[buf][mrow][kb + 16];
#pragma unroll
      for (int i = 0; i < 8; i++) { afrag[i] = p0[i]; afrag[8 + i] = p1[i]; }
    }
#pragma unroll
    for (int nt = 0; nt < NTILES; nt++) {
      v16h bfrag;
      const _Float16* q0 = &Bs[buf][nt * 16 + nlow][kb];
      const _Float16* q1 = &Bs[buf][nt * 16 + nlow][kb + 16];
#pragma unroll
      for (int i = 0; i < 8; i++) { bfrag[i] = q0[i]; bfrag[8 + i] = q1[i]; }
      acc[nt] = __builtin_amdgcn_wmma_f32_16x16x32_f16(
          false, afrag, false, bfrag, (short)0, acc[nt], false, false);
    }
    __syncthreads();                       // buffer free for next async overwrite
  }

  // ---- write-out per ISA C/D layout: VGPR v -> M = v (+8 for lanes>=16) ----
  const int mhi = (lane >> 4) * 8;
#pragma unroll
  for (int nt = 0; nt < NTILES; nt++) {
    int col = nt * 16 + nlow;
#pragma unroll
    for (int v = 0; v < 8; v++) {
      long row = rowBase + wave * 16 + mhi + v;
      if (row < N_NODES && col < ncols)
        Out[row * (long)ldout + colbase + col] = acc[nt][v];
    }
  }
}

// ---------------- per-node attention scores: s = h . a -----------------------
__global__ void gat_node_scores(const float* __restrict__ Hm, const float* __restrict__ a,
                                float* __restrict__ ssrc, float* __restrict__ sdst,
                                int N, int F, int nhead, int rowstride, int astride) {
  int idx = blockIdx.x * blockDim.x + threadIdx.x;
  if (idx >= N * nhead) return;
  int n = idx % N, h = idx / N;
  const float* row = Hm + (size_t)n * rowstride + h * F;
  const float* av  = a + (size_t)h * astride;
  float s0 = 0.f, s1 = 0.f;
  for (int j = 0; j < F; j++) { float x = row[j]; s0 += x * av[j]; s1 += x * av[F + j]; }
  ssrc[(size_t)h * N + n] = s0;
  sdst[(size_t)h * N + n] = s1;
}

// ---------------- segment max over dst (order-preserving uint atomics) -------
__global__ void gat_edge_max(const int* __restrict__ src, const int* __restrict__ dst,
                             const float* __restrict__ ssrc, const float* __restrict__ sdst,
                             unsigned int* __restrict__ mkey, int E, int nhead, int N) {
  int idx = blockIdx.x * blockDim.x + threadIdx.x;
  if (idx >= E * nhead) return;
  int e = idx % E, h = idx / E;
  int s = src[e], d = dst[e];
  float v = lrelu(ssrc[(size_t)h * N + s] + sdst[(size_t)h * N + d]);
  atomicMax(&mkey[(size_t)h * N + d], float_to_key(v));
}

// ---------------- segment sum of exp(e - m) ---------------------------------
__global__ void gat_edge_denom(const int* __restrict__ src, const int* __restrict__ dst,
                               const float* __restrict__ ssrc, const float* __restrict__ sdst,
                               const unsigned int* __restrict__ mkey, float* __restrict__ denom,
                               int E, int nhead, int N) {
  int idx = blockIdx.x * blockDim.x + threadIdx.x;
  if (idx >= E * nhead) return;
  int e = idx % E, h = idx / E;
  int s = src[e], d = dst[e];
  float v = lrelu(ssrc[(size_t)h * N + s] + sdst[(size_t)h * N + d]);
  float m = key_to_float(mkey[(size_t)h * N + d]);
  atomicAdd(&denom[(size_t)h * N + d], __expf(v - m));
}

// ---------------- layer-1 aggregation: wave per (edge, head), 64 feats -------
__global__ __launch_bounds__(256) void gat_aggregate1(
    const int* __restrict__ src, const int* __restrict__ dst,
    const float* __restrict__ ssrc, const float* __restrict__ sdst,
    const unsigned int* __restrict__ mkey, const float* __restrict__ denom,
    const float* __restrict__ Hm, float* __restrict__ outp, int E, int N) {
  int h = threadIdx.x >> 5;      // 8 waves = 8 heads
  int lane = threadIdx.x & 31;
  int e = blockIdx.x;
  if (e >= E) return;
  int s = src[e], d = dst[e];
  float v = lrelu(ssrc[(size_t)h * N + s] + sdst[(size_t)h * N + d]);
  float m = key_to_float(mkey[(size_t)h * N + d]);
  float alpha = __expf(v - m) / fmaxf(denom[(size_t)h * N + d], 1e-16f);
  const float2* hp = (const float2*)(Hm + (size_t)s * HCAT + h * NHID);
  float2 hv = hp[lane];
  float* op = outp + (size_t)d * HCAT + h * NHID + lane * 2;
  atomicAdd(op + 0, alpha * hv.x);
  atomicAdd(op + 1, alpha * hv.y);
}

// ---------------- ELU + convert to f16 for layer-2 GEMM ---------------------
__global__ void gat_elu_to_f16(const float* __restrict__ in, _Float16* __restrict__ out, long count) {
  long i = (long)blockIdx.x * blockDim.x + threadIdx.x;
  if (i >= count) return;
  float v = in[i];
  out[i] = (_Float16)(v > 0.f ? v : expm1f(v));
}

// ---------------- layer-2 aggregation: wave per edge, 40 feats --------------
__global__ __launch_bounds__(256) void gat_aggregate2(
    const int* __restrict__ src, const int* __restrict__ dst,
    const float* __restrict__ ssrc, const float* __restrict__ sdst,
    const unsigned int* __restrict__ mkey, const float* __restrict__ denom,
    const float* __restrict__ H2, float* __restrict__ outp, int E, int N) {
  int wave = threadIdx.x >> 5;
  int lane = threadIdx.x & 31;
  int e = blockIdx.x * 8 + wave;
  if (e >= E) return;
  int s = src[e], d = dst[e];
  float v = lrelu(ssrc[s] + sdst[d]);
  float m = key_to_float(mkey[d]);
  float alpha = __expf(v - m) / fmaxf(denom[d], 1e-16f);
  atomicAdd(&outp[(size_t)d * NCLASS + lane], alpha * H2[(size_t)s * NCLASS + lane]);
  if (lane < NCLASS - 32)
    atomicAdd(&outp[(size_t)d * NCLASS + 32 + lane], alpha * H2[(size_t)s * NCLASS + 32 + lane]);
}

// ---------------- log_softmax over 40 classes -------------------------------
__global__ void gat_log_softmax(const float* __restrict__ in, float* __restrict__ out, int N) {
  int n = blockIdx.x * blockDim.x + threadIdx.x;
  if (n >= N) return;
  const float* row = in + (size_t)n * NCLASS;
  float mx = row[0];
  for (int c = 1; c < NCLASS; c++) mx = fmaxf(mx, row[c]);
  float s = 0.f;
  for (int c = 0; c < NCLASS; c++) s += __expf(row[c] - mx);
  float ls = logf(s);
  float* orow = out + (size_t)n * NCLASS;
  for (int c = 0; c < NCLASS; c++) orow[c] = row[c] - mx - ls;
}

// ---------------------------------------------------------------------------
extern "C" void kernel_launch(void* const* d_in, const int* in_sizes, int n_in,
                              void* d_out, int out_size, void* d_ws, size_t ws_size,
                              hipStream_t stream) {
  (void)in_sizes; (void)n_in; (void)out_size; (void)ws_size;
  const float* x   = (const float*)d_in[0];
  const int*   el  = (const int*)  d_in[1];
  const float* W1  = (const float*)d_in[2];
  const float* a1  = (const float*)d_in[3];
  const float* b1  = (const float*)d_in[4];
  const float* W2  = (const float*)d_in[5];
  const float* a2  = (const float*)d_in[6];
  const float* b2  = (const float*)d_in[7];
  float* out = (float*)d_out;

  const int* src = el;
  const int* dst = el + N_EDGES;

  // workspace carve-out (~330 MB)
  char* ws = (char*)d_ws;
  size_t off = 0;
  auto carve = [&](size_t bytes) -> void* {
    void* p = ws + off;
    off = (off + bytes + 255) & ~(size_t)255;
    return p;
  };
  float*        Hm   = (float*)       carve((size_t)N_NODES * HCAT * 4);
  float*        out1 = (float*)       carve((size_t)N_NODES * HCAT * 4);
  _Float16*     Xh   = (_Float16*)    carve((size_t)N_NODES * NFEAT * 2);
  _Float16*     Hc   = (_Float16*)    carve((size_t)N_NODES * HCAT * 2);
  _Float16*     Wt1  = (_Float16*)    carve((size_t)NHEAD * NHID * NFEAT * 2);
  _Float16*     Wt2  = (_Float16*)    carve((size_t)64 * NFEAT * 2);
  float*        ss1  = (float*)       carve((size_t)NHEAD * N_NODES * 4);
  float*        sd1  = (float*)       carve((size_t)NHEAD * N_NODES * 4);
  unsigned int* mk1  = (unsigned int*)carve((size_t)NHEAD * N_NODES * 4);
  float*        dn1  = (float*)       carve((size_t)NHEAD * N_NODES * 4);
  float*        H2   = (float*)       carve((size_t)N_NODES * NCLASS * 4);
  float*        ss2  = (float*)       carve((size_t)N_NODES * 4);
  float*        sd2  = (float*)       carve((size_t)N_NODES * 4);
  unsigned int* mk2  = (unsigned int*)carve((size_t)N_NODES * 4);
  float*        dn2  = (float*)       carve((size_t)N_NODES * 4);
  float*        out2 = (float*)       carve((size_t)N_NODES * NCLASS * 4);

  const int T = 256;
  const int mblocks = (N_NODES + 127) / 128;   // 391

  // ---- one-time precision/layout prep ----
  gat_f32_to_f16<<<((long)N_NODES * NFEAT + T - 1) / T, T, 0, stream>>>(x, Xh, (long)N_NODES * NFEAT);
  gat_prep_w1<<<(NHEAD * NHID * NFEAT + T - 1) / T, T, 0, stream>>>(W1, Wt1);
  gat_prep_w2<<<(64 * NFEAT + T - 1) / T, T, 0, stream>>>(W2, Wt2);

  // ---- layer 1 ----
  gat_zero_md<<<(NHEAD * N_NODES + T - 1) / T, T, 0, stream>>>(mk1, dn1, NHEAD * N_NODES);
  gat_init_bias<<<((long)N_NODES * HCAT + T - 1) / T, T, 0, stream>>>(out1, b1, (long)N_NODES * HCAT, HCAT);
  gat_gemm_async_wmma<4><<<dim3(mblocks, NHEAD), T, 0, stream>>>(Xh, Wt1, Hm, HCAT, NHID, NHID);
  gat_node_scores<<<(NHEAD * N_NODES + T - 1) / T, T, 0, stream>>>(Hm, a1, ss1, sd1,
                                                                   N_NODES, NHID, NHEAD, HCAT, 2 * NHID);
  gat_edge_max<<<(NHEAD * N_EDGES + T - 1) / T, T, 0, stream>>>(src, dst, ss1, sd1, mk1,
                                                                N_EDGES, NHEAD, N_NODES);
  gat_edge_denom<<<(NHEAD * N_EDGES + T - 1) / T, T, 0, stream>>>(src, dst, ss1, sd1, mk1, dn1,
                                                                  N_EDGES, NHEAD, N_NODES);
  gat_aggregate1<<<N_EDGES, T, 0, stream>>>(src, dst, ss1, sd1, mk1, dn1, Hm, out1,
                                            N_EDGES, N_NODES);
  gat_elu_to_f16<<<((long)N_NODES * HCAT + T - 1) / T, T, 0, stream>>>(out1, Hc, (long)N_NODES * HCAT);

  // ---- layer 2 ----
  gat_zero_md<<<(N_NODES + T - 1) / T, T, 0, stream>>>(mk2, dn2, N_NODES);
  gat_init_bias<<<((long)N_NODES * NCLASS + T - 1) / T, T, 0, stream>>>(out2, b2, (long)N_NODES * NCLASS, NCLASS);
  gat_gemm_async_wmma<3><<<dim3(mblocks, 1), T, 0, stream>>>(Hc, Wt2, H2, NCLASS, 0, NCLASS);
  gat_node_scores<<<(N_NODES + T - 1) / T, T, 0, stream>>>(H2, a2, ss2, sd2,
                                                           N_NODES, NCLASS, 1, NCLASS, 2 * NCLASS);
  gat_edge_max<<<(N_EDGES + T - 1) / T, T, 0, stream>>>(src, dst, ss2, sd2, mk2,
                                                        N_EDGES, 1, N_NODES);
  gat_edge_denom<<<(N_EDGES + T - 1) / T, T, 0, stream>>>(src, dst, ss2, sd2, mk2, dn2,
                                                          N_EDGES, 1, N_NODES);
  gat_aggregate2<<<(N_EDGES + 7) / 8, T, 0, stream>>>(src, dst, ss2, sd2, mk2, dn2, H2, out2,
                                                      N_EDGES, N_NODES);
  gat_log_softmax<<<(N_NODES + T - 1) / T, T, 0, stream>>>(out2, out, N_NODES);
}
